// BnnResNet18_21397527069211
// MI455X (gfx1250) — 8-bit matrix operand order repaired and checked by reading, compiled, not run on hardware
//
#include <hip/hip_runtime.h>
#include <cstdint>


typedef __attribute__((ext_vector_type(8))) int v8i;

__global__ void k_zero(int8_t* __restrict__ p, int n) {
  int i = blockIdx.x * blockDim.x + threadIdx.x;
  if (i < n) p[i] = 0;
}

__global__ void k_conv1_pool(const float* __restrict__ x,
                             const float* __restrict__ w,
                             const float* __restrict__ bias,
                             const float* __restrict__ g, const float* __restrict__ be,
                             const float* __restrict__ mn, const float* __restrict__ vr,
                             float* __restrict__ outF, int8_t* __restrict__ out8) {
  int idx = blockIdx.x * blockDim.x + threadIdx.x;
  if (idx >= 512 * 16 * 16 * 64) return;
  int c  = idx & 63;
  int px = (idx >> 6) & 15;
  int py = (idx >> 10) & 15;
  int n  = idx >> 14;
  float s  = g[c] * rsqrtf(vr[c] + 1e-5f);
  float mb = mn[c], bb = be[c], b0 = bias[c];
  float best = -INFINITY;
  for (int wy = 0; wy < 3; ++wy) {
    int cy = 2 * py + wy - 1;
    if (cy < 0 || cy >= 32) continue;
    for (int wx = 0; wx < 3; ++wx) {
      int cx = 2 * px + wx - 1;
      if (cx < 0 || cx >= 32) continue;
      float acc = b0;
      for (int ci = 0; ci < 3; ++ci)
        for (int ky = 0; ky < 3; ++ky) {
          int iy = cy + ky - 1;
          if (iy < 0 || iy >= 32) continue;
          for (int kx = 0; kx < 3; ++kx) {
            int ix = cx + kx - 1;
            if (ix < 0 || ix >= 32) continue;
            acc += x[((n * 3 + ci) * 32 + iy) * 32 + ix] *
                   w[((c * 3 + ci) * 3 + ky) * 3 + kx];
          }
        }
      float v2 = fmaxf((acc - mb) * s + bb, 0.0f);
      best = fmaxf(best, v2);
    }
  }
  outF[idx] = best;
  out8[idx] = best > 0.0f ? (int8_t)1 : (int8_t)0;
}

__global__ void k_packw(const float* __restrict__ w, int8_t* __restrict__ bp,
                        int Cin, int Cout, int KHW, int Npad) {
  int Nt = Npad >> 4;
  int Kc = (Cin * KHW) >> 6;
  int total = Kc * Nt * 1024;
  int idx = blockIdx.x * blockDim.x + threadIdx.x;
  if (idx >= total) return;
  int b    = idx & 3;
  int r    = (idx >> 2) & 7;
  int lane = (idx >> 5) & 31;
  int frag = idx >> 10;
  int kc   = frag % Kc;
  int nt   = frag / Kc;
  int half = lane >> 4;
  int col  = lane & 15;
  int kin  = (r >> 1) * 16 + half * 8 + (r & 1) * 4 + b;
  int k    = kc * 64 + kin;
  int n    = nt * 16 + col;
  int8_t v = 0;
  if (n < Cout) {
    int tap = k / Cin;
    int ci  = k - tap * Cin;
    float f = w[((long)n * Cin + ci) * KHW + tap];
    v = f > 0.0f ? (int8_t)1 : (f < 0.0f ? (int8_t)-1 : (int8_t)0);
  }
  bp[idx] = v;
}

__global__ void __launch_bounds__(256)
k_bingemm_lds(const int8_t* __restrict__ A8, const int8_t* __restrict__ Bp,
              const int8_t* __restrict__ zpage, float* __restrict__ Craw,
              int Hin, int Win, int Cin, int Hout, int Wout, int Cout, int stride) {
  extern __shared__ int8_t ldsB[];
  const int Nt   = Cout >> 4;
  const int lane = threadIdx.x & 31;
  const int wave = threadIdx.x >> 5;
  const int nt   = blockIdx.x % Nt;
  const int mb   = (blockIdx.x / Nt) * 8 + wave;

  const int cpt = Cin >> 6;
  const int Kc  = 9 * cpt;

  {
    const int8_t* panel = Bp + (long)nt * Kc * 1024;
    const int bytes = Kc << 10;
    for (int i = threadIdx.x * 16; i < bytes; i += 256 * 16)
      *(int4*)(ldsB + i) = *(const int4*)(panel + i);
  }
  __syncthreads();

  const int l15   = lane & 15;
  const int half8 = (lane >> 4) << 3;

  int oxr[4], oyr[4], nr[4];
#pragma unroll
  for (int t = 0; t < 4; ++t) {
    int row = (mb << 6) + (t << 4) + l15;
    oxr[t] = row % Wout;
    int t2 = row / Wout;
    oyr[t] = t2 % Hout;
    nr[t]  = t2 / Hout;
  }

  v8i acc[4];
#pragma unroll
  for (int t = 0; t < 4; ++t) acc[t] = (v8i){0, 0, 0, 0, 0, 0, 0, 0};

  const int8_t* zp = zpage + half8;
  const int8_t* bp = ldsB + (lane << 5);

  for (int tap = 0; tap < 9; ++tap) {
    const int ky = tap / 3;
    const int kx = tap - ky * 3;
    const int8_t* ap[4];
#pragma unroll
    for (int t = 0; t < 4; ++t) {
      int iy = oyr[t] * stride + ky - 1;
      int ix = oxr[t] * stride + kx - 1;
      bool valid = (iy >= 0) & (iy < Hin) & (ix >= 0) & (ix < Win);
      const int8_t* p = A8 + ((long)(nr[t] * Hin + iy) * Win + ix) * (long)Cin + half8;
      ap[t] = valid ? p : zp;
      __builtin_prefetch(ap[t], 0, 1);
    }
    for (int cc = 0; cc < cpt; ++cc) {
      v8i b = *(const v8i*)bp;
      bp += 1024;
#pragma unroll
      for (int t = 0; t < 4; ++t) {
        const int8_t* p = ap[t] + (cc << 6);
        unsigned long long g0 = *(const unsigned long long*)(p);
        unsigned long long g1 = *(const unsigned long long*)(p + 16);
        unsigned long long g2 = *(const unsigned long long*)(p + 32);
        unsigned long long g3 = *(const unsigned long long*)(p + 48);
        v8i a;
        a[0] = (int)g0; a[1] = (int)(g0 >> 32);
        a[2] = (int)g1; a[3] = (int)(g1 >> 32);
        a[4] = (int)g2; a[5] = (int)(g2 >> 32);
        a[6] = (int)g3; a[7] = (int)(g3 >> 32);
        acc[t] = __builtin_amdgcn_wmma_i32_16x16x64_iu8(true, a, true, b, acc[t],
                                                        false, false);
      }
    }
  }
  float* outp = Craw + (long)((mb << 6) + half8) * Cout + (nt << 4) + l15;
#pragma unroll
  for (int t = 0; t < 4; ++t)
#pragma unroll
    for (int r = 0; r < 8; ++r)
      outp[(long)((t << 4) + r) * Cout] = (float)acc[t][r];
}

__global__ void k_bnrelu_bin(const float* __restrict__ raw,
                             const float* __restrict__ g, const float* __restrict__ be,
                             const float* __restrict__ mn, const float* __restrict__ vr,
                             int8_t* __restrict__ out8, long total, int C) {
  long idx = (long)blockIdx.x * blockDim.x + threadIdx.x;
  if (idx >= total) return;
  int c = (int)(idx % C);
  float s = g[c] * rsqrtf(vr[c] + 1e-5f);
  float v = fmaxf((raw[idx] - mn[c]) * s + be[c], 0.0f);
  out8[idx] = v > 0.0f ? (int8_t)1 : (int8_t)0;
}

__global__ void k_down(const float* __restrict__ inF, const float* __restrict__ w,
                       const float* __restrict__ g, const float* __restrict__ be,
                       const float* __restrict__ mn, const float* __restrict__ vr,
                       float* __restrict__ outF,
                       int Hin, int Win, int Cin, int Hout, int Wout, int Cout, int stride) {
  long total = (long)512 * Hout * Wout * Cout;
  long idx = (long)blockIdx.x * blockDim.x + threadIdx.x;
  if (idx >= total) return;
  int co = (int)(idx % Cout);
  long m = idx / Cout;
  int ox = (int)(m % Wout);
  long t2 = m / Wout;
  int oy = (int)(t2 % Hout);
  int n  = (int)(t2 / Hout);
  const float* src = inF + ((long)(n * Hin + oy * stride) * Win + ox * stride) * Cin;
  const float* wr  = w + (long)co * Cin;
  float acc = 0.0f;
  for (int ci = 0; ci < Cin; ++ci) acc += src[ci] * wr[ci];
  float s = g[co] * rsqrtf(vr[co] + 1e-5f);
  outF[idx] = (acc - mn[co]) * s + be[co];
}

__global__ void k_bnadd(const float* __restrict__ raw,
                        const float* __restrict__ g, const float* __restrict__ be,
                        const float* __restrict__ mn, const float* __restrict__ vr,
                        const float* __restrict__ idn,
                        float* __restrict__ outF, int8_t* __restrict__ out8,
                        long total, int C) {
  long idx = (long)blockIdx.x * blockDim.x + threadIdx.x;
  if (idx >= total) return;
  int c = (int)(idx % C);
  float s = g[c] * rsqrtf(vr[c] + 1e-5f);
  float v = (raw[idx] - mn[c]) * s + be[c] + idn[idx];
  v = fmaxf(v, 0.0f);
  outF[idx] = v;
  out8[idx] = v > 0.0f ? (int8_t)1 : (int8_t)0;
}

__global__ void k_avgbin(const float* __restrict__ inF, int8_t* __restrict__ a8) {
  int idx = blockIdx.x * blockDim.x + threadIdx.x;
  if (idx >= 512 * 512) return;
  int c = idx & 511;
  int n = idx >> 9;
  long base = (long)n * 4 * 512 + c;
  float m = (inF[base] + inF[base + 512] + inF[base + 1024] + inF[base + 1536]) * 0.25f;
  a8[idx] = m > 0.0f ? (int8_t)1 : (m < 0.0f ? (int8_t)-1 : (int8_t)0);
}

__global__ void __launch_bounds__(256)
k_fc_wmma(const int8_t* __restrict__ A8, const int8_t* __restrict__ Bp,
          const float* __restrict__ bias, float* __restrict__ out) {
  const int lane = threadIdx.x & 31;
  const int wave = threadIdx.x >> 5;
  const int mt   = blockIdx.x * 8 + wave;
  if (mt >= 32) return;
  const int half8 = (lane >> 4) << 3;
  const long rowbase = (long)((mt << 4) + (lane & 15)) * 512;
  v8i acc = {0, 0, 0, 0, 0, 0, 0, 0};
#pragma unroll
  for (int kc = 0; kc < 8; ++kc) {
    const int8_t* p = A8 + rowbase + (kc << 6) + half8;
    unsigned long long g0 = *(const unsigned long long*)(p);
    unsigned long long g1 = *(const unsigned long long*)(p + 16);
    unsigned long long g2 = *(const unsigned long long*)(p + 32);
    unsigned long long g3 = *(const unsigned long long*)(p + 48);
    v8i a;
    a[0] = (int)g0; a[1] = (int)(g0 >> 32);
    a[2] = (int)g1; a[3] = (int)(g1 >> 32);
    a[4] = (int)g2; a[5] = (int)(g2 >> 32);
    a[6] = (int)g3; a[7] = (int)(g3 >> 32);
    v8i b = *(const v8i*)(Bp + (kc << 10) + (lane << 5));
    acc = __builtin_amdgcn_wmma_i32_16x16x64_iu8(true, a, true, b, acc, false, false);
  }
  int col = lane & 15;
  if (col < 10) {
    float bb = bias[col];
#pragma unroll
    for (int r = 0; r < 8; ++r) {
      int row = (mt << 4) + r + ((lane >> 4) << 3);
      out[(long)row * 10 + col] = (float)acc[r] + bb;
    }
  }
}

extern "C" void kernel_launch(void* const* d_in, const int* in_sizes, int n_in,
                              void* d_out, int out_size, void* d_ws, size_t ws_size,
                              hipStream_t stream) {
  auto F = [&](int i) { return (const float*)d_in[i]; };

  const float* x    = F(0);
  const float* c1w  = F(1);
  const float* c1b  = F(2);
  const float *bn1g = F(3), *bn1b = F(4), *bn1m = F(5), *bn1v = F(6);
  struct Bn  { const float *g, *b, *m, *v; };
  struct Blk { const float* c1; Bn b1; const float* c2; Bn b2; const float* dw; Bn dbn; };
  Blk blk[8];
  const bool hasdown[8] = {false, false, true, false, true, false, true, false};
  int idx = 7;
  for (int i = 0; i < 8; ++i) {
    blk[i].c1 = F(idx++);
    blk[i].b1 = {F(idx), F(idx + 1), F(idx + 2), F(idx + 3)}; idx += 4;
    blk[i].c2 = F(idx++);
    blk[i].b2 = {F(idx), F(idx + 1), F(idx + 2), F(idx + 3)}; idx += 4;
    if (hasdown[i]) {
      blk[i].dw  = F(idx++);
      blk[i].dbn = {F(idx), F(idx + 1), F(idx + 2), F(idx + 3)}; idx += 4;
    } else {
      blk[i].dw = nullptr;
    }
  }
  const float* fcw = F(idx++);
  const float* fcb = F(idx++);

  char* ws = (char*)d_ws;
  size_t off = 0;
  auto carve = [&](size_t bytes) -> void* {
    void* p = ws + off;
    off += (bytes + 255) & ~(size_t)255;
    return p;
  };
  const size_t MAXACT = (size_t)8 * 1024 * 1024;
  float*  act0 = (float*)carve(MAXACT * 4);
  float*  act1 = (float*)carve(MAXACT * 4);
  float*  raw  = (float*)carve(MAXACT * 4);
  float*  idn  = (float*)carve((size_t)4 * 1024 * 1024 * 4);
  int8_t* a8   = (int8_t*)carve(MAXACT);
  int8_t* wb   = (int8_t*)carve((size_t)4608 * 512);
  int8_t* zpg  = (int8_t*)carve(1024);

  k_zero<<<4, 256, 0, stream>>>(zpg, 1024);

  {
    int total = 512 * 16 * 16 * 64;
    k_conv1_pool<<<total / 256, 256, 0, stream>>>(x, c1w, c1b, bn1g, bn1b, bn1m,
                                                  bn1v, act0, a8);
  }

  int H = 16, W = 16, C = 64;
  float* cur = act0;
  float* nxt = act1;
  const int couts[4] = {64, 128, 256, 512};
  int b = 0;
  for (int L = 0; L < 4; ++L) {
    for (int j = 0; j < 2; ++j, ++b) {
      Blk& B = blk[b];
      int stride = (j == 0 && L > 0) ? 2 : 1;
      int Cout = couts[L];
      int Ho = H / stride, Wo = W / stride;
      long M = (long)512 * Ho * Wo;
      long elems = M * Cout;
      int Nt = Cout >> 4;
      unsigned gemmBlocks = (unsigned)((M >> 9) * Nt);

      {
        long pbytes = (long)C * 9 * Cout;
        int lds1 = ((C * 9) >> 6) << 10;
        k_packw<<<(unsigned)((pbytes + 255) / 256), 256, 0, stream>>>(B.c1, wb, C, Cout, 9, Cout);
        k_bingemm_lds<<<gemmBlocks, 256, lds1, stream>>>(
            a8, wb, zpg, raw, H, W, C, Ho, Wo, Cout, stride);
        k_bnrelu_bin<<<(unsigned)((elems + 255) / 256), 256, 0, stream>>>(
            raw, B.b1.g, B.b1.b, B.b1.m, B.b1.v, a8, elems, Cout);
      }
      {
        long pbytes = (long)Cout * 9 * Cout;
        int lds2 = ((Cout * 9) >> 6) << 10;
        k_packw<<<(unsigned)((pbytes + 255) / 256), 256, 0, stream>>>(B.c2, wb, Cout, Cout, 9, Cout);
        k_bingemm_lds<<<gemmBlocks, 256, lds2, stream>>>(
            a8, wb, zpg, raw, Ho, Wo, Cout, Ho, Wo, Cout, 1);
      }
      const float* idptr = cur;
      if (B.dw) {
        k_down<<<(unsigned)((elems + 255) / 256), 256, 0, stream>>>(
            cur, B.dw, B.dbn.g, B.dbn.b, B.dbn.m, B.dbn.v, idn, H, W, C, Ho, Wo, Cout, stride);
        idptr = idn;
      }
      k_bnadd<<<(unsigned)((elems + 255) / 256), 256, 0, stream>>>(
          raw, B.b2.g, B.b2.b, B.b2.m, B.b2.v, idptr, nxt, a8, elems, Cout);

      float* t = cur; cur = nxt; nxt = t;
      H = Ho; W = Wo; C = Cout;
    }
  }

  k_avgbin<<<(512 * 512) / 256, 256, 0, stream>>>(cur, a8);
  k_packw<<<8192 / 256, 256, 0, stream>>>(fcw, wb, 512, 10, 1, 16);
  k_fc_wmma<<<4, 256, 0, stream>>>(a8, wb, fcb, (float*)d_out);
}
